// HeadwiseCrossViewInteraction_47553877902012
// MI455X (gfx1250) — matrix operand order repaired and checked by reading, not compiled, not run on hardware
//
#include <hip/hip_runtime.h>
#include <hip/hip_bf16.h>

typedef __attribute__((ext_vector_type(8)))  _Float16 v8h;
typedef __attribute__((ext_vector_type(16))) _Float16 v16h;
typedef __attribute__((ext_vector_type(8)))  float    v8f;

#define DHD 128
#define HN  16
#define VN  8
#define BT  16
#define BATCH 4096
#define AP  136   // padded LDS row stride in halves (272B -> bank-rotating b128 loads)

__device__ __forceinline__ v16h frag_cat(v8h lo, v8h hi) {
  return __builtin_shufflevector(lo, hi, 0,1,2,3,4,5,6,7,8,9,10,11,12,13,14,15);
}

// A fragment (16x32 f16, ISA layout): lanes 0-15: row=lane, K=kc+[0..7]&[16..23];
// lanes 16-31: row=lane-16, K=kc+[8..15]&[24..31]
__device__ __forceinline__ v16h load_a(const _Float16* base, int stride, int kt, int lane) {
  const int r  = lane & 15;
  const int hi = (lane >> 4) & 1;
  const _Float16* p = base + r * stride + kt * 32 + hi * 8;
  v8h lo = *(const v8h*)(p);
  v8h hh = *(const v8h*)(p + 16);
  return frag_cat(lo, hh);
}

// B fragment (32x16 f16): lane n holds column n (= row nt*16+n of row-major W);
// lanes 0-15 hold K=kc+[0..7]&[16..23], lanes 16-31 hold K=kc+[8..15]&[24..31]
__device__ __forceinline__ v16h load_b(const _Float16* w, int nt, int kt, int lane) {
  const int n  = lane & 15;
  const int hi = (lane >> 4) & 1;
  const _Float16* p = w + (size_t)(nt * 16 + n) * DHD + kt * 32 + hi * 8;
  v8h lo = *(const v8h*)(p);
  v8h hh = *(const v8h*)(p + 16);
  return frag_cat(lo, hh);
}

__device__ __forceinline__ v8f wmma16(v16h a, v16h b, v8f c) {
  return __builtin_amdgcn_wmma_f32_16x16x32_f16(false, a, false, b, (short)0, c, false, false);
}

// ---------------------------------------------------------------------------
// Kernel 1: per head, precompute Mt[d'][d] = sum_e Wq[e,d]*Wk[e,d'] (f16) and
// convert Wv/Wo to f16 in workspace. Tiny (16 blocks), runs once per launch.
// ---------------------------------------------------------------------------
__global__ __launch_bounds__(256) void prep_kernel(
    const float* __restrict__ Wq, const float* __restrict__ Wk,
    const float* __restrict__ Wv, const float* __restrict__ Wo,
    _Float16* __restrict__ Mt16, _Float16* __restrict__ Wv16,
    _Float16* __restrict__ Wo16)
{
  const int h = blockIdx.x;
  const size_t off = (size_t)h * DHD * DHD;
  const float* wq = Wq + off;
  const float* wk = Wk + off;
  for (int idx = threadIdx.x; idx < DHD * DHD; idx += 256) {
    const int dp = idx >> 7;      // d' (output feature of qt)
    const int d  = idx & (DHD-1); // d  (contraction with iso)
    float s = 0.f;
    for (int e = 0; e < DHD; ++e) s += wq[e * DHD + d] * wk[e * DHD + dp];
    Mt16[off + idx] = (_Float16)s;     // row-major [d'][d] => B-frag rows contiguous
  }
  for (int idx = threadIdx.x; idx < DHD * DHD; idx += 256) {
    Wv16[off + idx] = (_Float16)Wv[off + idx];
    Wo16[off + idx] = (_Float16)Wo[off + idx];
  }
}

// ---------------------------------------------------------------------------
// Kernel 2: fused attention. Block = (16 batches, 1 head), 128 threads (4 waves).
//   qt  = iso16 @ M          (WMMA, 8 N-tiles x 4 K-steps)
//   sc  = qt . x / sqrt(Dh) + log(pi+eps); attn = softmax_v(sc)
//   xb  = sum_v attn * x     (VALU, linearity trick removes v-projection)
//   ctx = xb16 @ Wv^T        (WMMA)
//   out = ctx16 @ Wo^T       (WMMA)
// ---------------------------------------------------------------------------
__global__ __launch_bounds__(128) void fused_kernel(
    const float* __restrict__ iso, const float* __restrict__ view,
    const float* __restrict__ pi,
    const _Float16* __restrict__ Mt16, const _Float16* __restrict__ Wv16,
    const _Float16* __restrict__ Wo16,
    float* __restrict__ outp, float* __restrict__ attnp)
{
  __shared__ __align__(16) _Float16 Xs[BT * VN * AP];  // view tokens f16
  __shared__ __align__(16) _Float16 As[BT * AP];       // iso f16
  __shared__ __align__(16) _Float16 Bx[BT * AP];       // xbar f16
  __shared__ __align__(16) _Float16 Cs[BT * AP];       // ctx f16
  __shared__ float qt[BT * DHD];
  __shared__ float sc[BT * VN];

  const int h    = blockIdx.y;
  const int b0   = blockIdx.x * BT;
  const int tid  = threadIdx.x;
  const int lane = tid & 31;
  const int wv   = tid >> 5;
  const int nt0  = 2 * wv, nt1 = 2 * wv + 1;

  // ---- stage iso + view tiles to LDS (f32 -> f16), fully coalesced ----
  for (int idx = tid; idx < BT * DHD; idx += 128) {
    const int m = idx >> 7, d = idx & (DHD-1);
    As[m * AP + d] = (_Float16)iso[((size_t)(b0 + m) * HN + h) * DHD + d];
  }
  for (int idx = tid; idx < BT * VN * DHD; idx += 128) {
    const int m  = idx >> 10;            // / (VN*DHD)
    const int vd = idx & 1023;
    const int v  = vd >> 7, d = vd & (DHD-1);
    Xs[(m * VN + v) * AP + d] =
        (_Float16)view[(((size_t)(b0 + m) * HN + h) * VN + v) * DHD + d];
  }
  __syncthreads();

  // ---- qt = iso @ M (via Mt = M^T row-major) ----
  {
    const _Float16* Mh = Mt16 + (size_t)h * DHD * DHD;
    v8f a0 = {}, a1 = {};
    #pragma unroll
    for (int kt = 0; kt < 4; ++kt) {
      const v16h af = load_a(As, AP, kt, lane);
      a0 = wmma16(af, load_b(Mh, nt0, kt, lane), a0);
      a1 = wmma16(af, load_b(Mh, nt1, kt, lane), a1);
    }
    const int n = lane & 15, hi = (lane >> 4) & 1;
    #pragma unroll
    for (int i = 0; i < 8; ++i) {
      qt[(i + 8 * hi) * DHD + nt0 * 16 + n] = a0[i];
      qt[(i + 8 * hi) * DHD + nt1 * 16 + n] = a1[i];
    }
  }
  __syncthreads();

  // ---- scores: one (b,v) per thread ----
  {
    const int b = tid >> 3, v = tid & 7;
    const float*    qr = &qt[b * DHD];
    const _Float16* xr = &Xs[(b * VN + v) * AP];
    float s = 0.f;
    #pragma unroll 8
    for (int d = 0; d < DHD; ++d) s += qr[d] * (float)xr[d];
    s *= 0.08838834764831845f;   // 1/sqrt(128)
    s += __logf(pi[((size_t)(b0 + b) * HN + h) * VN + v] + 1e-6f);
    sc[b * VN + v] = s;
  }
  __syncthreads();

  // ---- softmax over V=8, write attn output ----
  if (tid < BT) {
    const int b = tid;
    float mx = sc[b * VN];
    for (int v = 1; v < VN; ++v) mx = fmaxf(mx, sc[b * VN + v]);
    float ex[VN]; float sum = 0.f;
    for (int v = 0; v < VN; ++v) { ex[v] = __expf(sc[b * VN + v] - mx); sum += ex[v]; }
    const float inv = 1.f / sum;
    for (int v = 0; v < VN; ++v) {
      const float a = ex[v] * inv;
      sc[b * VN + v] = a;
      attnp[((size_t)(b0 + b) * HN + h) * VN + v] = a;
    }
  }
  __syncthreads();

  // ---- xbar[b,d] = sum_v attn[b,v] * x[b,v,d]  (linearity: replaces v-proj) ----
  for (int idx = tid; idx < BT * DHD; idx += 128) {
    const int b = idx >> 7, d = idx & (DHD-1);
    float s = 0.f;
    #pragma unroll
    for (int v = 0; v < VN; ++v) s += sc[b * VN + v] * (float)Xs[(b * VN + v) * AP + d];
    Bx[b * AP + d] = (_Float16)s;
  }
  __syncthreads();

  // ---- ctx = xbar @ Wv^T -> LDS f16 ----
  {
    const _Float16* Wvh = Wv16 + (size_t)h * DHD * DHD;
    v8f a0 = {}, a1 = {};
    #pragma unroll
    for (int kt = 0; kt < 4; ++kt) {
      const v16h af = load_a(Bx, AP, kt, lane);
      a0 = wmma16(af, load_b(Wvh, nt0, kt, lane), a0);
      a1 = wmma16(af, load_b(Wvh, nt1, kt, lane), a1);
    }
    const int n = lane & 15, hi = (lane >> 4) & 1;
    #pragma unroll
    for (int i = 0; i < 8; ++i) {
      Cs[(i + 8 * hi) * AP + nt0 * 16 + n] = (_Float16)a0[i];
      Cs[(i + 8 * hi) * AP + nt1 * 16 + n] = (_Float16)a1[i];
    }
  }
  __syncthreads();

  // ---- out = ctx @ Wo^T -> global f32 ----
  {
    const _Float16* Woh = Wo16 + (size_t)h * DHD * DHD;
    v8f a0 = {}, a1 = {};
    #pragma unroll
    for (int kt = 0; kt < 4; ++kt) {
      const v16h af = load_a(Cs, AP, kt, lane);
      a0 = wmma16(af, load_b(Woh, nt0, kt, lane), a0);
      a1 = wmma16(af, load_b(Woh, nt1, kt, lane), a1);
    }
    const int n = lane & 15, hi = (lane >> 4) & 1;
    #pragma unroll
    for (int i = 0; i < 8; ++i) {
      const int m = i + 8 * hi;
      outp[((size_t)(b0 + m) * HN + h) * DHD + nt0 * 16 + n] = a0[i];
      outp[((size_t)(b0 + m) * HN + h) * DHD + nt1 * 16 + n] = a1[i];
    }
  }
}

extern "C" void kernel_launch(void* const* d_in, const int* in_sizes, int n_in,
                              void* d_out, int out_size, void* d_ws, size_t ws_size,
                              hipStream_t stream) {
  const float* iso  = (const float*)d_in[0];
  const float* view = (const float*)d_in[1];
  const float* pi   = (const float*)d_in[2];
  const float* Wq   = (const float*)d_in[3];
  const float* Wk   = (const float*)d_in[4];
  const float* Wv   = (const float*)d_in[5];
  const float* Wo   = (const float*)d_in[6];

  _Float16* Mt16 = (_Float16*)d_ws;                    // [H][128][128] f16
  _Float16* Wv16 = Mt16 + (size_t)HN * DHD * DHD;      // [H][128][128] f16
  _Float16* Wo16 = Wv16 + (size_t)HN * DHD * DHD;      // [H][128][128] f16

  float* outp  = (float*)d_out;                        // [B,H,128]
  float* attnp = outp + (size_t)BATCH * HN * DHD;      // [B,H,8]

  prep_kernel<<<HN, 256, 0, stream>>>(Wq, Wk, Wv, Wo, Mt16, Wv16, Wo16);
  dim3 grid(BATCH / BT, HN);
  fused_kernel<<<grid, 128, 0, stream>>>(iso, view, pi, Mt16, Wv16, Wo16, outp, attnp);
}
